// IgLSTMRNNModule_53669911331241
// MI455X (gfx1250) — matrix operand order repaired and checked by reading, not compiled, not run on hardware
//
#include <hip/hip_runtime.h>
#include <hip/hip_bf16.h>

// ---------------- problem constants ----------------
#define B_   32
#define T_   1024
#define D_   768
#define H_   768
#define NG   3072            // 4*H (gate columns)
#define K2   1536            // H + D (folded K dimension)
#define NWG  48              // persistent workgroups (1 WGP each)
#define HU   16              // hidden units owned per WG  (NWG*HU == H)
#define TPB  256             // 8 waves of 32
#define KP   1552            // padded K stride (elements) for weight LDS rows
#define HP   784             // padded stride for h staging rows

// ---------------- LDS layout (bytes) ----------------
#define OFF_W   0
#define SZ_W    (64 * KP * 2)            // 198656  weights [64][KP] f16
#define OFF_AH  (OFF_W + SZ_W)
#define SZ_AH   (B_ * HP * 2)            // 50176   h_t staged [32][HP] f16
#define OFF_AX  (OFF_AH + SZ_AH)
#define SZ_AX   (B_ * D_ * 2)            // 49152   x_t staged [32][768] f16 (TDM dest)
#define OFF_G   (OFF_AX + SZ_AX)
#define SZ_G    (4 * B_ * 16 * 4)        // 8192    gate tiles [4][32][16] f32
#define OFF_BS  (OFF_G + SZ_G)
#define SZ_BS   (64 * 4)
#define SMEM_BYTES (OFF_BS + SZ_BS)      // 306432 < 320KB WGP LDS

typedef __attribute__((ext_vector_type(16))) _Float16 v16h;
typedef __attribute__((ext_vector_type(8)))  _Float16 v8h;
typedef __attribute__((ext_vector_type(8)))  float    v8f;
typedef __attribute__((ext_vector_type(4)))  unsigned int u32x4;
typedef __attribute__((ext_vector_type(8)))  int i32x8;
typedef __attribute__((ext_vector_type(4)))  int i32x4;

#if __has_builtin(__builtin_amdgcn_tensor_load_to_lds) && \
    __has_builtin(__builtin_amdgcn_s_wait_tensorcnt)
#define HAVE_TDM 1
#else
#define HAVE_TDM 0
#endif

// ---------------- init: convert to f16, fuse bias, reset state ----------------
__global__ void lstm_init(const float* __restrict__ hid,
                          const float* __restrict__ Wih,
                          const float* __restrict__ Whh,
                          const float* __restrict__ bih,
                          const float* __restrict__ bhh,
                          _Float16* __restrict__ wcomb,   // [NG][K2]
                          _Float16* __restrict__ x16,     // [B][T][D]
                          float*    __restrict__ biasc,   // [NG]
                          _Float16* __restrict__ hbuf,    // [2][B][H]
                          unsigned* __restrict__ bar)
{
    size_t tid    = (size_t)blockIdx.x * blockDim.x + threadIdx.x;
    size_t stride = (size_t)gridDim.x * blockDim.x;

    // fused weight matrix: row n = [ W_hh[n][0..H) | W_ih[n][0..D) ]
    for (size_t i = tid; i < (size_t)NG * K2; i += stride) {
        size_t n = i / K2, k = i % K2;
        float w = (k < H_) ? Whh[n * H_ + k] : Wih[n * D_ + (k - H_)];
        wcomb[i] = (_Float16)w;
    }
    for (size_t i = tid; i < (size_t)B_ * T_ * D_; i += stride)
        x16[i] = (_Float16)hid[i];
    for (size_t i = tid; i < (size_t)NG; i += stride)
        biasc[i] = bih[i] + bhh[i];
    for (size_t i = tid; i < (size_t)2 * B_ * H_; i += stride)
        hbuf[i] = (_Float16)0.f;
    if (tid == 0) *bar = 0u;
}

#if HAVE_TDM
// TDM: async 2D tile load, 32 rows x 768 f16, row stride T_*D_ elements.
// D# bit packing per CDNA5 ISA ch.8 (group0: count/lds/global/type=2;
// group1: data_size, tensor dims, tile dims, strides).
// clang-23 / therock-10.0 builtin form: 6 args (g0, g1, g2, g3, g4, cpol).
__device__ inline void tdm_load_x(const _Float16* gsrc, void* ldsdst) {
    unsigned long long ga = (unsigned long long)(uintptr_t)gsrc;
    unsigned lds = (unsigned)(uintptr_t)ldsdst;  // generic addr low 32 = LDS offset
    u32x4 g0;
    g0.x = 1u;                                           // count=1 (valid), user mode
    g0.y = lds;                                          // lds_addr
    g0.z = (unsigned)(ga & 0xFFFFFFFFu);                 // global_addr[31:0]
    g0.w = (unsigned)((ga >> 32) & 0x01FFFFFFu)          // global_addr[56:32]
         | 0x80000000u;                                  // type=2 ("image")
    i32x8 g1;
    g1[0] = (int)(1u << 16);          // data_size=1 -> 2 bytes; no multicast/pad
    g1[1] = (int)((unsigned)D_ << 16);// tensor_dim0[15:0]=768
    g1[2] = (int)((unsigned)B_ << 16);// tensor_dim0 hi=0 | tensor_dim1[15:0]=32
    g1[3] = (int)((unsigned)D_ << 16);// tensor_dim1 hi=0 | tile_dim0=768
    g1[4] = B_;                       // tile_dim1=32, tile_dim2=0
    g1[5] = (int)(T_ * D_);           // tensor_dim0_stride[31:0] = 786432
    g1[6] = 0;                        // stride0 hi | tensor_dim1_stride lo
    g1[7] = 0;
    i32x4 gz4 = {0, 0, 0, 0};
    i32x8 gz8 = {0, 0, 0, 0, 0, 0, 0, 0};
    __builtin_amdgcn_tensor_load_to_lds(g0, g1, gz4, gz4, gz8, 0);
}
#endif

// ---------------- persistent WMMA scan kernel ----------------
__global__ void __launch_bounds__(TPB) lstm_scan(
    const _Float16* __restrict__ wcomb,   // [NG][K2] f16
    const _Float16* __restrict__ x16,     // [B][T][D] f16
    const float*    __restrict__ biasc,   // [NG]
    _Float16*       __restrict__ hbuf,    // [2][B][H] f16 ping-pong
    unsigned*       __restrict__ bar,     // monotonic barrier counter
    float*          __restrict__ out)     // [B][T][H] f32
{
    extern __shared__ __align__(16) char smem[];
    _Float16* Wl = (_Float16*)(smem + OFF_W);
    _Float16* Ah = (_Float16*)(smem + OFF_AH);
    _Float16* Ax = (_Float16*)(smem + OFF_AX);
    float*    Gl = (float*)   (smem + OFF_G);
    float*    bs = (float*)   (smem + OFF_BS);

    const int wg  = blockIdx.x;          // 0..47
    const int hu0 = wg * HU;             // first owned hidden unit
    const int tid = threadIdx.x;

    // ---- preload owned weight slice into LDS: Wl[nl][k], nl = gate*16 + j ----
    for (int v = tid; v < 64 * (K2 / 8); v += TPB) {
        int nl = v / (K2 / 8);
        int k8 = (v % (K2 / 8)) * 8;
        int ng = (nl >> 4) * H_ + hu0 + (nl & 15);   // global gate column
        *(v8h*)(Wl + nl * KP + k8) = *(const v8h*)(wcomb + (size_t)ng * K2 + k8);
    }
    if (tid < 64) {
        int ng = (tid >> 4) * H_ + hu0 + (tid & 15);
        bs[tid] = biasc[ng];
    }

    const int lane = tid & 31;
    const int wv   = tid >> 5;           // wave 0..7
    const int mt   = wv >> 2;            // M tile 0..1 (batch 0-15 / 16-31)
    const int gt   = wv & 3;             // gate 0..3 (i,f,g,o)
    const int lh   = lane >> 4;          // lane half
    const int lm   = lane & 15;

    float creg[2] = {0.f, 0.f};          // cell state, fixed thread->(m,j) map

#if HAVE_TDM
    if (wv == 0) tdm_load_x(x16, Ax);    // prefetch x_0 via Tensor Data Mover
#endif
    __syncthreads();

    for (int t = 0; t < T_; ++t) {
        const _Float16* hcur  = hbuf + (size_t)(t & 1) * (B_ * H_);
        _Float16*       hnext = hbuf + (size_t)((t + 1) & 1) * (B_ * H_);

        // ---- stage h_t into LDS (x_t already arriving via TDM) ----
        for (int v = tid; v < B_ * (H_ / 8); v += TPB) {
            int m = v / (H_ / 8);
            int k8 = (v % (H_ / 8)) * 8;
            *(v8h*)(Ah + m * HP + k8) = *(const v8h*)(hcur + m * H_ + k8);
        }
#if HAVE_TDM
        if (wv == 0) __builtin_amdgcn_s_wait_tensorcnt(0);
#else
        for (int v = tid; v < B_ * (D_ / 8); v += TPB) {
            int m = v / (D_ / 8);
            int k8 = (v % (D_ / 8)) * 8;
            *(v8h*)(Ax + m * D_ + k8) =
                *(const v8h*)(x16 + ((size_t)m * T_ + t) * D_ + k8);
        }
#endif
        __syncthreads();

        // ---- one 16x16 tile per wave; 2 x 24 k-tiles, software pipelined ----
        v8f acc0 = {};
        v8f acc1 = {};
        const _Float16* brow = Wl + (gt * 16 + lm) * KP;   // B: N col = lm
        {
            const _Float16* arh = Ah + (mt * 16 + lm) * HP;      // h region rows
            const _Float16* arx = Ax + (mt * 16 + lm) * D_;      // x region rows

            // lambda over one 24-k-tile region; kb continues in B via 'koff'
            auto run24 = [&](const _Float16* ar, int koff) {
                v8h alo0 = *(const v8h*)(ar + 0 + lh * 16);
                v8h ahi0 = *(const v8h*)(ar + 8 + lh * 16);
                v16h b0  = *(const v16h*)(brow + koff + lh * 16);
                v8h alo1, ahi1; v16h b1;
                #pragma unroll 4
                for (int kt = 0; kt < 24; kt += 2) {
                    int kb1 = (kt + 1) * 32;
                    alo1 = *(const v8h*)(ar + kb1 + lh * 16);
                    ahi1 = *(const v8h*)(ar + kb1 + 8 + lh * 16);
                    b1   = *(const v16h*)(brow + koff + kb1 + lh * 16);
                    v16h a0 = __builtin_shufflevector(alo0, ahi0,
                            0,1,2,3,4,5,6,7,8,9,10,11,12,13,14,15);
                    acc0 = __builtin_amdgcn_wmma_f32_16x16x32_f16(
                            false, a0, false, b0, (short)0, acc0, false, false);
                    int kb2 = (kt + 2 < 24) ? (kt + 2) * 32 : 0;
                    alo0 = *(const v8h*)(ar + kb2 + lh * 16);
                    ahi0 = *(const v8h*)(ar + kb2 + 8 + lh * 16);
                    b0   = *(const v16h*)(brow + koff + kb2 + lh * 16);
                    v16h a1 = __builtin_shufflevector(alo1, ahi1,
                            0,1,2,3,4,5,6,7,8,9,10,11,12,13,14,15);
                    acc1 = __builtin_amdgcn_wmma_f32_16x16x32_f16(
                            false, a1, false, b1, (short)0, acc1, false, false);
                }
            };
            run24(arh, 0);        // K in [0, 768)   : h @ W_hh^T
            run24(arx, D_);       // K in [768, 1536): x @ W_ih^T
        }
        v8f acc = acc0 + acc1;

        // ---- spill gate tile to LDS: D layout VGPR r -> M = r + lh*8 ----
        #pragma unroll
        for (int r = 0; r < 8; ++r)
            Gl[(gt * B_ + mt * 16 + lh * 8 + r) * 16 + lm] = acc[r];
        __syncthreads();          // all waves done reading Ax, Gl complete

#if HAVE_TDM
        // kick TDM for x_{t+1}; overlaps elementwise + global barrier
        if (wv == 0 && t + 1 < T_) tdm_load_x(x16 + (size_t)(t + 1) * D_, Ax);
#endif

        // ---- elementwise LSTM cell: 512 cells, 2 per thread ----
        #pragma unroll
        for (int rr = 0; rr < 2; ++rr) {
            int idx = tid + TPB * rr;            // 0..511
            int m = idx >> 4;                    // batch row
            int j = idx & 15;                    // owned hidden unit
            float iv = Gl[(0 * B_ + m) * 16 + j] + bs[ 0 + j];
            float fv = Gl[(1 * B_ + m) * 16 + j] + bs[16 + j];
            float gv = Gl[(2 * B_ + m) * 16 + j] + bs[32 + j];
            float ov = Gl[(3 * B_ + m) * 16 + j] + bs[48 + j];
            iv = 1.f / (1.f + __expf(-iv));
            fv = 1.f / (1.f + __expf(-fv));
            gv = tanhf(gv);
            ov = 1.f / (1.f + __expf(-ov));
            float c = fv * creg[rr] + iv * gv;
            creg[rr] = c;
            float h = ov * tanhf(c);
            hnext[m * H_ + hu0 + j] = (_Float16)h;
            out[((size_t)m * T_ + t) * H_ + hu0 + j] = h;
        }

        // ---- grid-wide barrier (monotonic counter, deterministic) ----
        __threadfence();
        __syncthreads();
        if (tid == 0) {
            __hip_atomic_fetch_add(bar, 1u, __ATOMIC_RELEASE,
                                   __HIP_MEMORY_SCOPE_AGENT);
            unsigned target = (unsigned)NWG * (unsigned)(t + 1);
            while (__hip_atomic_load(bar, __ATOMIC_ACQUIRE,
                                     __HIP_MEMORY_SCOPE_AGENT) < target) {
                __builtin_amdgcn_s_sleep(2);
            }
        }
        __syncthreads();
        __threadfence();   // acquire: see other WGPs' h writes
    }
}

// ---------------- launcher ----------------
extern "C" void kernel_launch(void* const* d_in, const int* in_sizes, int n_in,
                              void* d_out, int out_size, void* d_ws, size_t ws_size,
                              hipStream_t stream) {
    (void)in_sizes; (void)n_in; (void)out_size;

    const float* hid = (const float*)d_in[0];  // [B,T,D]
    const float* Wih = (const float*)d_in[1];  // [4H,D]
    const float* Whh = (const float*)d_in[2];  // [4H,H]
    const float* bih = (const float*)d_in[3];  // [4H]
    const float* bhh = (const float*)d_in[4];  // [4H]
    float* out = (float*)d_out;                // [B,T,H]

    // workspace carve-up (all 16B aligned)
    char* ws = (char*)d_ws;
    size_t off = 0;
    _Float16* wcomb = (_Float16*)(ws + off); off += (size_t)NG * K2 * 2;        // 9.4 MB
    _Float16* x16   = (_Float16*)(ws + off); off += (size_t)B_ * T_ * D_ * 2;   // 48 MB
    float*    biasc = (float*)   (ws + off); off += (size_t)NG * 4;
    _Float16* hbuf  = (_Float16*)(ws + off); off += (size_t)2 * B_ * H_ * 2;
    unsigned* bar   = (unsigned*)(ws + off); off += 128;
    if (off > ws_size) { /* workspace too small; nothing sane to do */ }

    lstm_init<<<2048, 256, 0, stream>>>(hid, Wih, Whh, bih, bhh,
                                        wcomb, x16, biasc, hbuf, bar);

    (void)hipFuncSetAttribute((const void*)lstm_scan,
                              hipFuncAttributeMaxDynamicSharedMemorySize,
                              SMEM_BYTES);
    lstm_scan<<<NWG, TPB, SMEM_BYTES, stream>>>(wcomb, x16, biasc, hbuf, bar, out);
}
